// NRIConv_42666205118900
// MI455X (gfx1250) — matrix operand order repaired and checked by reading, not compiled, not run on hardware
//
#include <hip/hip_runtime.h>

typedef __attribute__((ext_vector_type(16))) _Float16 v16h;
typedef __attribute__((ext_vector_type(8)))  _Float16 v8h;
typedef __attribute__((ext_vector_type(8)))  float    v8f;

#define NB 256     // nodes
#define NE 65280   // 256*255 edges
#define F  64      // in features
#define H  64      // hidden / out channels
#define ET 2       // edge types
#define BATCH 4

// ---------------------------------------------------------------------------
// prep: per-node first-layer projections (factorized first edge-MLP layer),
// stored in f16 so the per-edge hidden build is packed-f16 math.
//   Xs[i][b][n][k] = sum_f inputs[b,n,f] * W1[i][f][k]
//   Xr[i][b][n][k] = sum_f inputs[b,n,f] * W1[i][64+f][k] + b1[i][k]
// ---------------------------------------------------------------------------
__global__ void nri_prep_node(const float* __restrict__ inputs,
                              const float* __restrict__ W1,
                              const float* __restrict__ b1,
                              _Float16* __restrict__ Xs,
                              _Float16* __restrict__ Xr) {
  const int blk = blockIdx.x;          // b*256 + n
  const int t = threadIdx.x;           // 0..63 output channel
  __shared__ float xf[F];
  xf[t] = inputs[blk * F + t];
  __syncthreads();
  const int b = blk >> 8, n = blk & 255;
#pragma unroll
  for (int i = 0; i < ET; ++i) {
    float ss = 0.f, sr = 0.f;
#pragma unroll 8
    for (int f = 0; f < F; ++f) {
      const float x = xf[f];
      ss += x * W1[(i * 2 * F + f) * H + t];
      sr += x * W1[(i * 2 * F + F + f) * H + t];
    }
    Xs[(((i * BATCH) + b) * NB + n) * H + t] = (_Float16)ss;
    Xr[(((i * BATCH) + b) * NB + n) * H + t] = (_Float16)(sr + b1[i * H + t]);
  }
}

// ---------------------------------------------------------------------------
// Pre-pack W2 into this kernel's WMMA B-fragment lane order (f16, contiguous k), so the main kernel
// loads each 32x16 B fragment with two contiguous b128 loads per lane.
// W2p index: ((((i*4 + nt)*2 + ks)*32 + lane)*16 + kk)
//   element = W2[i][ ks*32 + (lane>=16?16:0) + kk ][ nt*16 + (lane&15) ]
// ---------------------------------------------------------------------------
__global__ void nri_w2_pack(const float* __restrict__ W2,
                            _Float16* __restrict__ W2p) {
  const int idx = blockIdx.x * blockDim.x + threadIdx.x;
  if (idx >= ET * 4 * 2 * 32 * 16) return;
  const int kk   = idx & 15;
  const int lane = (idx >> 4) & 31;
  const int ks   = (idx >> 9) & 1;
  const int nt   = (idx >> 10) & 3;
  const int i    = idx >> 12;
  const int k = ks * 32 + ((lane >> 4) * 16) + kk;
  const int n = nt * 16 + (lane & 15);
  W2p[idx] = (_Float16)W2[(i * H + k) * H + n];
}

// ---------------------------------------------------------------------------
// main: one workgroup per (batch b, receiver r). 4 waves of 32 (wave32).
// Software-pipelined per-wave tile loop with double-buffered LDS:
//   hidden = relu(Xs[s] + Xr[r])  (packed f16, stored in A-fragment order)
//   msg    = hidden @ W2[i]       (v_wmma_f32_16x16x32_f16, K=64 in 2 steps)
//   agg   += relu(msg + b2[i]) * rel_type[b,e,i]  (segment sum, ds_add_f32)
// Then fused per-node output MLP.
// ---------------------------------------------------------------------------
__global__ void __launch_bounds__(128) nri_edge_main(
    const float* __restrict__ inputs, const float* __restrict__ rel_type,
    const _Float16* __restrict__ Xs, const _Float16* __restrict__ Xr,
    const _Float16* __restrict__ W2p, const float* __restrict__ b2,
    const float* __restrict__ Wo1, const float* __restrict__ bo1,
    const float* __restrict__ Wo2, const float* __restrict__ bo2,
    float* __restrict__ out) {
  const int b = blockIdx.x >> 8;
  const int r = blockIdx.x & 255;
  const int lane = threadIdx.x & 31;
  const int wave = threadIdx.x >> 5;

  // hidden tiles, A-fragment order, double buffered: [wave][buf][ks][lane][16]
  __shared__ __align__(16) _Float16 hidA[4][2][2][32][16];
  __shared__ float agg[H];
  __shared__ float p1s[H];

  if (threadIdx.x < H) agg[threadIdx.x] = 0.f;
  __builtin_prefetch(rel_type + (b * NE + r * 255) * ET, 0, 1);
  __syncthreads();

  const int ncol = lane & 15;      // B/C column index
  const int hi16 = lane >> 4;      // upper-half-lane flag
  // writer role for hidden build: 2 lanes per row, 32 K's per lane
  const int wm  = lane >> 1;       // row (edge within tile)
  const int wks = lane & 1;        // which 32-K block this lane produces

  for (int i = 0; i < ET; ++i) {
    const _Float16* XsBase = Xs + (((i * BATCH) + b) * NB) * H + wks * 32;

    // ---- B fragments: pre-packed, 2 x b128 contiguous loads per fragment
    v16h Bf[4][2];
#pragma unroll
    for (int nt = 0; nt < 4; ++nt)
#pragma unroll
      for (int ks = 0; ks < 2; ++ks)
        Bf[nt][ks] =
            *(const v16h*)(W2p + ((((i * 4) + nt) * 2 + ks) * 32 + lane) * 16);

    float bias[4];
#pragma unroll
    for (int nt = 0; nt < 4; ++nt) bias[nt] = b2[i * H + nt * 16 + ncol];

    // ---- receiver projection chunk for this lane's writer role (32 halves)
    v8h xrv[4];
    {
      const _Float16* xrp =
          Xr + (((i * BATCH) + b) * NB + r) * H + wks * 32;
#pragma unroll
      for (int q = 0; q < 4; ++q) xrv[q] = *(const v8h*)(xrp + q * 8);
    }

    // ---- pipeline prologue: load sender chunks for this wave's first tile
    v8h xsv[4];
    {
      const int j = wave * 16 + wm;
      int s = (j < r) ? j : j + 1;
      if (s > 255) s = 255;
      const _Float16* xsp = XsBase + s * H;
#pragma unroll
      for (int q = 0; q < 4; ++q) xsv[q] = *(const v8h*)(xsp + q * 8);
    }

    for (int tile = wave; tile < 16; tile += 4) {  // 4 iterations per wave
      const int buf = (tile >> 2) & 1;

      // ---- hidden = relu(Xs[s] + Xr[r]) from prefetched regs, A-frag order
      {
        v8h hh[4];
        const v8h z = {};
#pragma unroll
        for (int q = 0; q < 4; ++q)
          hh[q] = __builtin_elementwise_max(xsv[q] + xrv[q], z);
        // K offs 0..7 -> (lane=wm, halves 0..7); 8..15 -> (wm, 8..15);
        // 16..23 -> (wm+16, 0..7); 24..31 -> (wm+16, 8..15)
        *(v8h*)&hidA[wave][buf][wks][wm][0]      = hh[0];
        *(v8h*)&hidA[wave][buf][wks][wm][8]      = hh[1];
        *(v8h*)&hidA[wave][buf][wks][wm + 16][0] = hh[2];
        *(v8h*)&hidA[wave][buf][wks][wm + 16][8] = hh[3];
      }

      // ---- prefetch next tile's sender chunks (latency hidden under WMMAs)
      {
        const int tn = tile + 4;
        if (tn < 16) {
          const int j = tn * 16 + wm;
          int s = (j < r) ? j : j + 1;
          if (s > 255) s = 255;
          const _Float16* xsp = XsBase + s * H;
#pragma unroll
          for (int q = 0; q < 4; ++q) xsv[q] = *(const v8h*)(xsp + q * 8);
        }
      }

      // per-wave LDS executes in order; fence the scheduler only.
      // Single barrier per iteration: orders store(buf)@t before load(buf)@t,
      // and (via the chain of barriers) load(buf)@t before store(buf)@t+2.
      __builtin_amdgcn_wave_barrier();

      // ---- A fragments: single contiguous 32B LDS read each
      const v16h A0 = *(const v16h*)&hidA[wave][buf][0][lane][0];
      const v16h A1 = *(const v16h*)&hidA[wave][buf][1][lane][0];

      // ---- per-row rel_type weights (D layout: VGPR j <-> row j / 8+j)
      float w[8];
#pragma unroll
      for (int j2 = 0; j2 < 8; ++j2) {
        const int eloc = tile * 16 + hi16 * 8 + j2;
        w[j2] = (eloc < 255)
                    ? rel_type[(b * NE + r * 255 + eloc) * ET + i]
                    : 0.f;
      }

      // ---- WMMA second layer + fused epilogue (bias, relu, weight, segsum)
#pragma unroll
      for (int nt = 0; nt < 4; ++nt) {
        v8f c = {};
        c = __builtin_amdgcn_wmma_f32_16x16x32_f16(
            false, A0, false, Bf[nt][0], (short)0, c, false, false);
        c = __builtin_amdgcn_wmma_f32_16x16x32_f16(
            false, A1, false, Bf[nt][1], (short)0, c, false, false);
        float sacc = 0.f;
#pragma unroll
        for (int j2 = 0; j2 < 8; ++j2)
          sacc += fmaxf(c[j2] + bias[nt], 0.f) * w[j2];
        atomicAdd(&agg[nt * 16 + ncol], sacc);   // ds_add_f32
      }
    }
  }
  __syncthreads();   // all waves' agg atomics complete

  // ---- fused output MLP: pred = relu(relu([x, agg]@Wo1+bo1)@Wo2+bo2)
  if (threadIdx.x < H) {
    const int t = threadIdx.x;
    const float* xin = inputs + (b * NB + r) * F;
    float p = bo1[t];
#pragma unroll 8
    for (int c0 = 0; c0 < F; ++c0) p += xin[c0] * Wo1[c0 * H + t];
#pragma unroll 8
    for (int c0 = 0; c0 < H; ++c0) p += agg[c0] * Wo1[(F + c0) * H + t];
    p1s[t] = fmaxf(p, 0.f);
  }
  __syncthreads();
  if (threadIdx.x < H) {
    const int t = threadIdx.x;
    const float* xin = inputs + (b * NB + r) * F;
    float q = bo2[t];
#pragma unroll 8
    for (int c0 = 0; c0 < H; ++c0) q += p1s[c0] * Wo2[c0 * H + t];
    out[(b * NB + r) * H + t] = xin[t] + fmaxf(q, 0.f);
  }
}

// ---------------------------------------------------------------------------
extern "C" void kernel_launch(void* const* d_in, const int* in_sizes, int n_in,
                              void* d_out, int out_size, void* d_ws,
                              size_t ws_size, hipStream_t stream) {
  (void)in_sizes; (void)n_in; (void)out_size; (void)ws_size;
  const float* inputs   = (const float*)d_in[0];
  const float* rel_type = (const float*)d_in[1];
  // d_in[2]/d_in[3] (rel_rec/rel_send) unused: NRI edge structure is analytic.
  const float* W1  = (const float*)d_in[4];
  const float* b1  = (const float*)d_in[5];
  const float* W2  = (const float*)d_in[6];
  const float* b2  = (const float*)d_in[7];
  const float* Wo1 = (const float*)d_in[8];
  const float* bo1 = (const float*)d_in[9];
  const float* Wo2 = (const float*)d_in[10];
  const float* bo2 = (const float*)d_in[11];
  float* out = (float*)d_out;

  // workspace (f16): Xs 256KB | Xr 256KB | W2p 16KB
  _Float16* Xs  = (_Float16*)d_ws;
  _Float16* Xr  = Xs + (size_t)ET * BATCH * NB * H;
  _Float16* W2p = Xr + (size_t)ET * BATCH * NB * H;

  nri_prep_node<<<BATCH * NB, 64, 0, stream>>>(inputs, W1, b1, Xs, Xr);
  nri_w2_pack<<<(ET * 4 * 2 * 32 * 16 + 255) / 256, 256, 0, stream>>>(W2, W2p);
  nri_edge_main<<<BATCH * NB, 128, 0, stream>>>(inputs, rel_type, Xs, Xr, W2p,
                                                b2, Wo1, bo1, Wo2, bo2, out);
}
